// FDSM_40295383171692
// MI455X (gfx1250) — compile-verified, matrix operand order checked by reading, not run on hardware
//
#include <hip/hip_runtime.h>

// Sizes from the reference
#define BB   8
#define CCH  128      // channels C
#define FFN  4        // filters F
#define HH   128
#define WWD  128
#define WFQ  65
#define HWQ  16384    // H*W

typedef __attribute__((ext_vector_type(16))) _Float16 v16h;
typedef __attribute__((ext_vector_type(8)))  float    v8f;

struct c2 { float x, y; };
__device__ inline c2 cmul(c2 a, c2 b){ c2 r; r.x = a.x*b.x - a.y*b.y; r.y = a.x*b.y + a.y*b.x; return r; }
__device__ inline c2 cadd(c2 a, c2 b){ c2 r; r.x = a.x+b.x; r.y = a.y+b.y; return r; }
__device__ inline c2 csub(c2 a, c2 b){ c2 r; r.x = a.x-b.x; r.y = a.y-b.y; return r; }
__device__ inline c2 cconj(c2 a){ c2 r; r.x = a.x; r.y = -a.y; return r; }

// ---------------------------------------------------------------------------
// Kernel A: fold GroupNorm affine + duplicated-concat into one 128x128 weight
// (f16 for WMMA), fold beta into bias, zero the pooled accumulator.
// ---------------------------------------------------------------------------
__global__ void fdsm_prep(const float* __restrict__ agg_w, const float* __restrict__ agg_b,
                          const float* __restrict__ gamma, const float* __restrict__ beta,
                          _Float16* __restrict__ weff, float* __restrict__ beff,
                          float* __restrict__ pooled)
{
    int t = threadIdx.x;
    for (int i = t; i < BB*CCH; i += blockDim.x) pooled[i] = 0.f;
    for (int i = t; i < CCH*CCH; i += blockDim.x) {
        int o = i >> 7, c = i & 127;
        float v = agg_w[o*2*CCH + c] * gamma[c] + agg_w[o*2*CCH + CCH + c] * gamma[CCH + c];
        weff[o*CCH + c] = (_Float16)v;
    }
    if (t < CCH) {
        float s = agg_b[t];
        for (int c = 0; c < 2*CCH; ++c) s += agg_w[t*2*CCH + c] * beta[c];
        beff[t] = s;
    }
}

// ---------------------------------------------------------------------------
// Kernel B: GN statistics. cat([x,x]) => 16 distinct (b,group) stats over x,
// each 8 channels * HW = 131072 elements.
// ---------------------------------------------------------------------------
__global__ void fdsm_gnstats(const float* __restrict__ x,
                             float* __restrict__ mean, float* __restrict__ rstd)
{
    __shared__ float s1[256], s2[256];
    int g = blockIdx.x;                 // 0..127  (b*16 + grp)
    int b = g >> 4, grp = g & 15;
    const float* p = x + ((size_t)(b*CCH + grp*8)) * HWQ;
    float a = 0.f, q = 0.f;
    for (int i = threadIdx.x; i < 8*HWQ; i += 256) { float v = p[i]; a += v; q += v*v; }
    s1[threadIdx.x] = a; s2[threadIdx.x] = q; __syncthreads();
    for (int off = 128; off > 0; off >>= 1) {
        if (threadIdx.x < off) { s1[threadIdx.x] += s1[threadIdx.x+off]; s2[threadIdx.x] += s2[threadIdx.x+off]; }
        __syncthreads();
    }
    if (threadIdx.x == 0) {
        float m  = s1[0] * (1.f / (8.f*HWQ));
        float vr = s2[0] * (1.f / (8.f*HWQ)) - m*m;
        mean[g] = m; rstd[g] = rsqrtf(vr + 1e-5f);
    }
}

// ---------------------------------------------------------------------------
// Build a 16-bit A/B WMMA fragment from LDS. ISA 7.12.2 layout:
// lane<16: row=lane, K = kb..kb+7 and kb+16..kb+23 (kb = k0)
// lane>=16: row=lane-16, same with kb = k0+8.
// ---------------------------------------------------------------------------
__device__ inline v16h load_frag(const _Float16* __restrict__ p)
{
    v16h r;
#pragma unroll
    for (int e = 0; e < 8; ++e) r[e] = p[e];
#pragma unroll
    for (int e = 0; e < 8; ++e) r[8+e] = p[16+e];
    return r;
}

// ---------------------------------------------------------------------------
// Kernel C: fused  normalize -> f16 -> GEMM(WMMA) -> +bias -> ReLU -> column
// sums (global average pool accumulation). Tile: 64 pixels x 128 out-channels,
// K = 128. 2048 blocks (256 tiles per batch). 8 waves: wave = (mt*2 + nh),
// mt in 0..3 picks 16 rows, nh picks 64 columns (4 WMMA tiles).
// ---------------------------------------------------------------------------
__global__ void __launch_bounds__(256) fdsm_gemm_pool(
    const float* __restrict__ x, const float* __restrict__ mean, const float* __restrict__ rstd,
    const _Float16* __restrict__ weff, const float* __restrict__ beff,
    float* __restrict__ pooled)
{
    const int STRIDE = 132;                       // pad rows: 264B, 8B aligned
    __shared__ _Float16 As[64*STRIDE];            // [pixel][k]
    __shared__ _Float16 Bs[128*STRIDE];           // [outCh][k]
    __shared__ float    pacc[128];

    int tid = threadIdx.x;
    int g   = blockIdx.x;                         // 0..2047
    int b   = g >> 8;
    int p0  = (g & 255) << 6;                     // 64-pixel tile base

    if (tid < 128) pacc[tid] = 0.f;

    for (int i = tid; i < CCH*CCH; i += 256) {    // weights
        Bs[(i >> 7)*STRIDE + (i & 127)] = weff[i];
    }
    const float* mb = mean + b*16;
    const float* rb = rstd + b*16;
    for (int i = tid; i < 64*CCH; i += 256) {     // A: normalize on the fly
        int c = i >> 6, p = i & 63;
        float v = x[((size_t)(b*CCH + c))*HWQ + p0 + p];
        int gg = c >> 3;
        v = (v - mb[gg]) * rb[gg];
        As[p*STRIDE + c] = (_Float16)v;
    }
    __syncthreads();

    int wv = tid >> 5, lane = tid & 31;
    int r  = lane & 15, lh = lane >> 4;
    int mt = wv >> 1, nh = wv & 1;

    v8f acc[4];
    v8f zero = {0.f,0.f,0.f,0.f,0.f,0.f,0.f,0.f};
#pragma unroll
    for (int nt = 0; nt < 4; ++nt) acc[nt] = zero;

#pragma unroll
    for (int k0 = 0; k0 < 128; k0 += 32) {
        int kb = k0 + lh*8;
        v16h afrag = load_frag(As + (mt*16 + r)*STRIDE + kb);
#pragma unroll
        for (int nt = 0; nt < 4; ++nt) {
            int ntile = nh*4 + nt;
            v16h bfrag = load_frag(Bs + (ntile*16 + r)*STRIDE + kb);
            acc[nt] = __builtin_amdgcn_wmma_f32_16x16x32_f16(
                false, afrag, false, bfrag, (short)0, acc[nt], false, false);
        }
    }

    // bias + ReLU + reduce over the 16 M-rows of this wave's tiles
#pragma unroll
    for (int nt = 0; nt < 4; ++nt) {
        int n = (nh*4 + nt)*16 + r;               // output channel (lane%16 = N)
        float bia = beff[n];
        float s = 0.f;
#pragma unroll
        for (int j = 0; j < 8; ++j) {             // m = j + 8*lh
            float v = acc[nt][j] + bia;
            s += (v > 0.f) ? v : 0.f;
        }
        s += __shfl_xor(s, 16, 32);               // combine both M-halves
        if (lh == 0) atomicAdd(&pacc[n], s);
    }
    __syncthreads();
    if (tid < 128) atomicAdd(&pooled[b*CCH + tid], pacc[tid]);
}

// ---------------------------------------------------------------------------
// Kernel D: per-batch filter weights = softmax(pooled/HW @ wg_w^T + wg_b).
// 32 threads = one wave; softmax across 4-lane groups via shfl_xor.
// ---------------------------------------------------------------------------
__global__ void fdsm_weights(const float* __restrict__ pooled, const float* __restrict__ wg_w,
                             const float* __restrict__ wg_b, float* __restrict__ wts)
{
    int t = threadIdx.x;                          // 0..31 == b*4 + f
    int b = t >> 2, f = t & 3;
    float lg = wg_b[f];
    for (int c = 0; c < CCH; ++c)
        lg += pooled[b*CCH + c] * (1.f/(float)HWQ) * wg_w[f*CCH + c];
    float m = lg;
    m = fmaxf(m, __shfl_xor(m, 1, 32));
    m = fmaxf(m, __shfl_xor(m, 2, 32));
    float e = __expf(lg - m);
    float s = e;
    s += __shfl_xor(s, 1, 32);
    s += __shfl_xor(s, 2, 32);
    wts[t] = e / s;
}

// ---------------------------------------------------------------------------
// Kernel E: fused spectral pipeline per (b,c) image.
//   fwd 2D FFT (radix-2 DIF, in-place, output bit-reversed)
//   pointwise: P = X^2 * Geff   (Hermitian-projection identity: ff=rfft2(irfft2)
//              is the identity on interior columns, an (h,H-h) conj-average on
//              w=0 and w=W/2; mirrored w>W/2 uses conj(G) -- fully pointwise)
//   inv 2D FFT (radix-2 DIT, bit-reversed input -> natural), scale 2^-21,
//   add residual_weight * x.
// ---------------------------------------------------------------------------
__global__ void __launch_bounds__(256) fdsm_fft(
    const float* __restrict__ x, const float* __restrict__ filt,
    const float* __restrict__ wts, const float* __restrict__ rw,
    float* __restrict__ out)
{
    extern __shared__ char smem[];
    const int PAD = 129;
    c2* S  = (c2*)smem;            // [128][129] complex
    c2* tw = S + 128*PAD;          // [64] twiddles exp(-2pi i j/128)

    int tid = threadIdx.x;
    int bc  = blockIdx.x;          // 0..1023
    int b   = bc >> 7, c = bc & 127;
    const size_t base = (size_t)bc * HWQ;

    if (tid < 64) {
        float ang = -3.14159265358979323846f * (float)tid / 64.f;
        float sn, cs; __sincosf(ang, &sn, &cs);
        c2 t; t.x = cs; t.y = sn; tw[tid] = t;
    }
    for (int i = tid; i < HWQ; i += 256) {
        c2 v; v.x = x[base + i]; v.y = 0.f;
        S[(i >> 7)*PAD + (i & 127)] = v;
    }
    __syncthreads();

    // ---- forward DIF along W (rows) ----
    for (int s = 7; s >= 1; --s) {
        int half = 1 << (s-1);
        for (int t = tid; t < 8192; t += 256) {
            int row = t >> 6, bt = t & 63;
            int grp = bt >> (s-1), j = bt & (half-1);
            int i0 = (grp << s) + j, i1 = i0 + half;
            c2 w = tw[j << (7 - s)];
            c2* pr = S + row*PAD;
            c2 u = pr[i0], v = pr[i1];
            pr[i0] = cadd(u, v);
            pr[i1] = cmul(csub(u, v), w);
        }
        __syncthreads();
    }
    // ---- forward DIF along H (cols) ----
    for (int s = 7; s >= 1; --s) {
        int half = 1 << (s-1);
        for (int t = tid; t < 8192; t += 256) {
            int col = t & 127, ct = t >> 7;
            int grp = ct >> (s-1), j = ct & (half-1);
            int i0 = (grp << s) + j, i1 = i0 + half;
            c2 w = tw[j << (7 - s)];
            c2 u = S[i0*PAD + col], v = S[i1*PAD + col];
            S[i0*PAD + col] = cadd(u, v);
            S[i1*PAD + col] = cmul(csub(u, v), w);
        }
        __syncthreads();
    }

    // ---- pointwise spectral combine (storage is bit-reversed in both axes) ----
    float w0 = wts[b*FFN+0], w1 = wts[b*FFN+1], w2 = wts[b*FFN+2], w3 = wts[b*FFN+3];
    const float*  fbase   = filt + (size_t)c * (HH*WFQ*2);      // f=0 slab for channel c
    const size_t  fstride = (size_t)CCH * HH * WFQ * 2;         // stride between filters
    for (int idx = tid; idx < HWQ; idx += 256) {
        int i = idx >> 7, jj = idx & 127;
        int h = (int)(__brev((unsigned)i)  >> 25);              // rev7
        int w = (int)(__brev((unsigned)jj) >> 25);
        c2 X = S[i*PAD + jj];

        int fh, fw; bool cj = false, edge = false;
        if (w >= 65)               { fh = (128 - h) & 127; fw = 128 - w; cj = true; }
        else if (w == 0 || w == 64){ fh = h; fw = w; edge = true; }
        else                       { fh = h; fw = w; }

        size_t off = (size_t)(fh*WFQ + fw) * 2;
        const float* q0 = fbase + off;
        __builtin_prefetch(q0 + 2*WFQ, 0, 0);                   // stream next filter row
        const float* q1 = q0 + fstride;
        const float* q2 = q1 + fstride;
        const float* q3 = q2 + fstride;
        c2 G;
        G.x = w0*q0[0] + w1*q1[0] + w2*q2[0] + w3*q3[0];
        G.y = w0*q0[1] + w1*q1[1] + w2*q2[1] + w3*q3[1];
        if (cj) G = cconj(G);
        if (edge) {
            int h2 = (128 - h) & 127;
            size_t off2 = (size_t)(h2*WFQ + fw) * 2;
            const float* r0 = fbase + off2;
            const float* r1 = r0 + fstride;
            const float* r2 = r1 + fstride;
            const float* r3 = r2 + fstride;
            c2 G2;
            G2.x = w0*r0[0] + w1*r1[0] + w2*r2[0] + w3*r3[0];
            G2.y = w0*r0[1] + w1*r1[1] + w2*r2[1] + w3*r3[1];
            G2 = cconj(G2);
            G.x = 0.5f*(G.x + G2.x); G.y = 0.5f*(G.y + G2.y);
        }
        c2 X2 = cmul(X, X);
        S[i*PAD + jj] = cmul(X2, G);
    }
    __syncthreads();

    // ---- inverse DIT along H (cols), conj twiddles, unnormalized ----
    for (int s = 1; s <= 7; ++s) {
        int half = 1 << (s-1);
        for (int t = tid; t < 8192; t += 256) {
            int col = t & 127, ct = t >> 7;
            int grp = ct >> (s-1), j = ct & (half-1);
            int i0 = (grp << s) + j, i1 = i0 + half;
            c2 w = cconj(tw[j << (7 - s)]);
            c2 u = S[i0*PAD + col];
            c2 v = cmul(S[i1*PAD + col], w);
            S[i0*PAD + col] = cadd(u, v);
            S[i1*PAD + col] = csub(u, v);
        }
        __syncthreads();
    }
    // ---- inverse DIT along W (rows) ----
    for (int s = 1; s <= 7; ++s) {
        int half = 1 << (s-1);
        for (int t = tid; t < 8192; t += 256) {
            int row = t >> 6, bt = t & 63;
            int grp = bt >> (s-1), j = bt & (half-1);
            int i0 = (grp << s) + j, i1 = i0 + half;
            c2 w = cconj(tw[j << (7 - s)]);
            c2* pr = S + row*PAD;
            c2 u = pr[i0];
            c2 v = cmul(pr[i1], w);
            pr[i0] = cadd(u, v);
            pr[i1] = csub(u, v);
        }
        __syncthreads();
    }

    // ortho^2 scaling folded: 1/(128^2 * 128) = 2^-21; plus residual.
    float rwv = rw[0];
    const float scale = 1.f / 2097152.f;
    for (int i = tid; i < HWQ; i += 256) {
        out[base + i] = S[(i >> 7)*PAD + (i & 127)].x * scale + rwv * x[base + i];
    }
}

// ---------------------------------------------------------------------------
extern "C" void kernel_launch(void* const* d_in, const int* in_sizes, int n_in,
                              void* d_out, int out_size, void* d_ws, size_t ws_size,
                              hipStream_t stream)
{
    (void)in_sizes; (void)n_in; (void)out_size; (void)ws_size;
    const float* x     = (const float*)d_in[0];
    const float* gamma = (const float*)d_in[1];
    const float* beta  = (const float*)d_in[2];
    const float* agg_w = (const float*)d_in[3];
    const float* agg_b = (const float*)d_in[4];
    const float* wg_w  = (const float*)d_in[5];
    const float* wg_b  = (const float*)d_in[6];
    const float* filt  = (const float*)d_in[7];
    const float* rw    = (const float*)d_in[8];
    float* out = (float*)d_out;

    char* ws = (char*)d_ws;
    _Float16* weff = (_Float16*)ws;                 // 128*128*2 = 32768 B
    float* beff    = (float*)(ws + 32768);          // 512 B
    float* mean    = (float*)(ws + 33280);          // 512 B
    float* rstd    = (float*)(ws + 33792);          // 512 B
    float* pooled  = (float*)(ws + 34304);          // 4096 B
    float* wts     = (float*)(ws + 38400);          // 128 B

    fdsm_prep     <<<1,    256, 0, stream>>>(agg_w, agg_b, gamma, beta, weff, beff, pooled);
    fdsm_gnstats  <<<128,  256, 0, stream>>>(x, mean, rstd);
    fdsm_gemm_pool<<<2048, 256, 0, stream>>>(x, mean, rstd, weff, beff, pooled);
    fdsm_weights  <<<1,    32,  0, stream>>>(pooled, wg_w, wg_b, wts);

    size_t fftLds = (size_t)(128*129 + 64) * sizeof(c2);   // ~132.6 KB dynamic LDS
    fdsm_fft      <<<1024, 256, fftLds, stream>>>(x, filt, wts, rw, out);
}
